// MultiHeadAttention_51737176048450
// MI455X (gfx1250) — matrix operand order repaired and checked by reading, not compiled, not run on hardware
//
#include <hip/hip_runtime.h>

// ---------------------------------------------------------------------------
// MI455X (gfx1250, wave32) attention block.
// - all matmuls: v_wmma_f32_16x16x32_bf16
// - GEMM tiles: double-buffered global_load_async_to_lds_b128 (ASYNCcnt
//   in-order completion => s_wait_asynccnt 4 overlaps next-tile loads with
//   current-tile WMMAs)
// - flash attention: online softmax with wave32 __shfl_xor reductions
// ---------------------------------------------------------------------------

typedef __bf16 v16bf  __attribute__((ext_vector_type(16)));
typedef __bf16 bf16x8 __attribute__((ext_vector_type(8)));
typedef __bf16 bf16x4 __attribute__((ext_vector_type(4)));
typedef float  v8f    __attribute__((ext_vector_type(8)));
typedef float  f32x4  __attribute__((ext_vector_type(4)));

#define B_    2
#define S_    2048
#define H_    4096
#define NH_   32
#define NKV_  8
#define HD_   128
#define NREP_ 4
#define SCALE_ 0.08838834764831845f   // 128^-0.5

__device__ __forceinline__ __bf16 f2bf(float f) {
  unsigned u = __builtin_bit_cast(unsigned, f);
  unsigned r = (u + 0x7FFFu + ((u >> 16) & 1u)) >> 16;
  unsigned short s = (unsigned short)r;
  return __builtin_bit_cast(__bf16, s);
}
__device__ __forceinline__ float bf2f(__bf16 b) {
  unsigned short s = __builtin_bit_cast(unsigned short, b);
  unsigned u = ((unsigned)s) << 16;
  return __builtin_bit_cast(float, u);
}
__device__ __forceinline__ v16bf join8(bf16x8 lo, bf16x8 hi) {
  return __builtin_shufflevector(lo, hi, 0,1,2,3,4,5,6,7,8,9,10,11,12,13,14,15);
}
__device__ __forceinline__ v8f wmma_bf16(v16bf a, v16bf b, v8f c) {
  return __builtin_amdgcn_wmma_f32_16x16x32_bf16(false, a, false, b, (short)0, c,
                                                 false, false);
}

// gfx1250 async copy global -> LDS (16B per lane), tracked by ASYNCcnt.
__device__ __forceinline__ void async_copy16(unsigned lds_off, const void* gaddr) {
  asm volatile("global_load_async_to_lds_b128 %0, %1, off"
               :: "v"(lds_off), "v"(gaddr) : "memory");
}
__device__ __forceinline__ void wait_async0() {
  asm volatile("s_wait_asynccnt 0" ::: "memory");
}
__device__ __forceinline__ void wait_async4() {   // first 4 outstanding done
  asm volatile("s_wait_asynccnt 4" ::: "memory");
}
__device__ __forceinline__ unsigned lds_off_of(const void* p) {
  return (unsigned)(uintptr_t)p;   // shared aperture: low 32 bits = LDS offset
}

// ---------------------------------------------------------------------------
// fp32 -> bf16 bulk conversion (4 elems/thread)
// ---------------------------------------------------------------------------
__global__ __launch_bounds__(256) void cvt_f32_bf16(
    const float* __restrict__ in, __bf16* __restrict__ out, long n4)
{
  const long i = (long)blockIdx.x * blockDim.x + threadIdx.x;
  if (i < n4) {
    f32x4 v = *(const f32x4*)&in[i * 4];
    bf16x4 o;
    o[0] = f2bf(v[0]); o[1] = f2bf(v[1]); o[2] = f2bf(v[2]); o[3] = f2bf(v[3]);
    *(bf16x4*)&out[i * 4] = o;
  }
}

// ---------------------------------------------------------------------------
// C[M,N] = A[M,K] @ W[N,K]^T (+bias). A,W bf16. 128x128 tile, 8 waves each
// computing 32x64. Double-buffered async-to-LDS tile pipeline.
// ---------------------------------------------------------------------------
template <bool HAS_BIAS, bool OUT_F32>
__global__ __launch_bounds__(256) void gemm_bf16(
    const __bf16* __restrict__ A, const __bf16* __restrict__ W,
    const float* __restrict__ bias,
    __bf16* __restrict__ outB, float* __restrict__ outF,
    int M, int N, int K)
{
  __shared__ __attribute__((aligned(16))) __bf16 As[2][128 * 48];  // 12KB each
  __shared__ __attribute__((aligned(16))) __bf16 Bs[2][128 * 48];

  const int tid  = threadIdx.x;
  const int wave = tid >> 5, lane = tid & 31;
  const int lh = lane >> 4, l15 = lane & 15;
  const int wm = wave >> 1, wn = wave & 1;
  const long m0 = (long)blockIdx.y * 128, n0 = (long)blockIdx.x * 128;

  const unsigned asB0 = lds_off_of(As[0]), asB1 = lds_off_of(As[1]);
  const unsigned bsB0 = lds_off_of(Bs[0]), bsB1 = lds_off_of(Bs[1]);
  // this thread's two 16B chunks of each 128x32 tile
  const int rA0 = tid >> 2,         cA0 = (tid & 3) * 8;
  const int rA1 = (tid + 256) >> 2, cA1 = ((tid + 256) & 3) * 8;
  const unsigned o0 = (unsigned)(rA0 * 96 + cA0 * 2);
  const unsigned o1 = (unsigned)(rA1 * 96 + cA1 * 2);
  const char* Ab = (const char*)A;
  const char* Wb = (const char*)W;

  auto issue = [&](int buf, int k0) {
    const unsigned ab = buf ? asB1 : asB0;
    const unsigned bb = buf ? bsB1 : bsB0;
    async_copy16(ab + o0, Ab + ((m0 + rA0) * (long)K + k0 + cA0) * 2);
    async_copy16(ab + o1, Ab + ((m0 + rA1) * (long)K + k0 + cA1) * 2);
    async_copy16(bb + o0, Wb + ((n0 + rA0) * (long)K + k0 + cA0) * 2);
    async_copy16(bb + o1, Wb + ((n0 + rA1) * (long)K + k0 + cA1) * 2);
  };

  v8f c[2][4] = {};
  issue(0, 0);                       // prologue: tile 0 in flight
  int cur = 0;

  for (int k0 = 0; k0 < K; k0 += 32) {
    __syncthreads();                 // readers of buf cur^1 (tile t-1) done
    if (k0 + 32 < K) {
      issue(cur ^ 1, k0 + 32);       // stream tile t+1 during compute of t
      wait_async4();                 // in-order completion: tile t landed
    } else {
      wait_async0();
    }
    __syncthreads();                 // all waves' tile t landed

    const __bf16* as = As[cur];
    const __bf16* bs = Bs[cur];
    v16bf a[2], b[4];
#pragma unroll
    for (int i = 0; i < 2; ++i) {
      const int r = wm * 32 + i * 16 + l15;
      bf16x8 lo = *(const bf16x8*)&as[r * 48 + lh * 8];
      bf16x8 hi = *(const bf16x8*)&as[r * 48 + 16 + lh * 8];
      a[i] = join8(lo, hi);
    }
#pragma unroll
    for (int j = 0; j < 4; ++j) {
      const int r = wn * 64 + j * 16 + l15;
      bf16x8 lo = *(const bf16x8*)&bs[r * 48 + lh * 8];
      bf16x8 hi = *(const bf16x8*)&bs[r * 48 + 16 + lh * 8];
      b[j] = join8(lo, hi);
    }
#pragma unroll
    for (int i = 0; i < 2; ++i)
#pragma unroll
      for (int j = 0; j < 4; ++j)
        c[i][j] = wmma_bf16(a[i], b[j], c[i][j]);
    cur ^= 1;
  }

  // branch-free epilogue: C frag element (m = r + 8*lh, n = l15)
#pragma unroll
  for (int j = 0; j < 4; ++j) {
    const long n = n0 + wn * 64 + j * 16 + l15;
    float bv = 0.0f;
    if (HAS_BIAS) bv = bias[n];
#pragma unroll
    for (int i = 0; i < 2; ++i)
#pragma unroll
      for (int r = 0; r < 8; ++r) {
        const long m = m0 + wm * 32 + i * 16 + r + 8 * lh;
        const float v = c[i][j][r] + bv;
        if (OUT_F32) outF[m * N + n] = v;
        else         outB[m * N + n] = f2bf(v);
      }
  }
}

// ---------------------------------------------------------------------------
// RoPE + reorder [B,S,heads*HD] -> [B,heads,S,HD].
// ---------------------------------------------------------------------------
__global__ __launch_bounds__(64) void rope_reorder(
    const __bf16* __restrict__ in, const float* __restrict__ cosp,
    const float* __restrict__ sinp, __bf16* __restrict__ out,
    int heads, int doRope)
{
  int idx = blockIdx.x;
  const int h = idx % heads; idx /= heads;
  const int s = idx % S_;    const int b = idx / S_;
  const int d = threadIdx.x;                      // 0..63 (HD/2)
  const long ibase = ((long)(b * S_ + s) * heads + h) * HD_;
  const long obase = ((long)(b * heads + h) * S_ + s) * HD_;
  if (doRope) {
    const float x1 = bf2f(in[ibase + 2 * d]);
    const float x2 = bf2f(in[ibase + 2 * d + 1]);
    const float cc = cosp[s * (HD_ / 2) + d];
    const float ss = sinp[s * (HD_ / 2) + d];
    out[obase + d]           = f2bf(x1 * cc - x2 * ss);
    out[obase + HD_ / 2 + d] = f2bf(x1 * ss + x2 * cc);
  } else {
    out[obase + d]           = in[ibase + d];
    out[obase + HD_ / 2 + d] = in[ibase + HD_ / 2 + d];
  }
}

// ---------------------------------------------------------------------------
// GQA flash attention: 32 q rows per workgroup, 64-key tiles, online softmax.
// Q/K tiles via async-to-LDS; V staged manually (in-LDS transpose).
// Softmax: 8 lanes per row, wave32 __shfl_xor reductions.
// ---------------------------------------------------------------------------
__global__ __launch_bounds__(256) void flash_attn(
    const __bf16* __restrict__ Q, const __bf16* __restrict__ Kk,
    const __bf16* __restrict__ V, __bf16* __restrict__ Out)
{
  __shared__ __attribute__((aligned(16))) __bf16 Qs[32 * 128];    // 8 KB
  __shared__ __attribute__((aligned(16))) __bf16 Ks[64 * 128];    // 16 KB
  __shared__ __attribute__((aligned(16))) __bf16 Vts[128 * 64];   // 16 KB [d][key]
  __shared__ __attribute__((aligned(16))) float  Ss[32 * 64];     // 8 KB
  __shared__ __attribute__((aligned(16))) __bf16 Pb[32 * 64];     // 4 KB
  __shared__ float rmax[32], rsum[32], rscale[32];

  const int tid  = threadIdx.x;
  const int wave = tid >> 5, lane = tid & 31;
  const int lh = lane >> 4, l15 = lane & 15;
  const int mt = wave & 1;           // 16-row q tile
  const int wq = wave >> 1;          // 0..3: key tile (QK) / d-base/32 (PV)

  int idx = blockIdx.x;
  const int qt = idx % (S_ / 32); idx /= (S_ / 32);
  const int h  = idx % NH_;       const int b = idx / NH_;
  const int kvh = h / NREP_;

  const long qbase  = ((long)(b * NH_ + h) * S_ + qt * 32) * HD_;
  const long kbase0 = ((long)(b * NKV_ + kvh) * S_) * HD_;

  const unsigned qsBase = lds_off_of(Qs);
  const unsigned ksBase = lds_off_of(Ks);

  // Q tile 32x128 via async copies (2 x 16B per thread)
#pragma unroll
  for (int ci = 0; ci < 2; ++ci) {
    const int ch = tid + ci * 256;
    const int row = ch >> 4, col = (ch & 15) * 8;
    async_copy16(qsBase + (unsigned)(row * 256 + col * 2),
                 (const char*)Q + (qbase + row * HD_ + col) * 2);
  }
  if (tid < 32) { rmax[tid] = -1e30f; rsum[tid] = 0.0f; }

  v8f o[2] = {};

  for (int kt = 0; kt < S_ / 64; ++kt) {
    __syncthreads();
    const long kb = kbase0 + (long)kt * 64 * HD_;
    // K tile 64x128 via async (4 x 16B per thread)
#pragma unroll
    for (int ci = 0; ci < 4; ++ci) {
      const int ch = tid + ci * 256;
      const int row = ch >> 4, col = (ch & 15) * 8;
      async_copy16(ksBase + (unsigned)(row * 256 + col * 2),
                   (const char*)Kk + (kb + row * HD_ + col) * 2);
    }
    // V tile -> transposed LDS (element scatter)
#pragma unroll
    for (int ci = 0; ci < 4; ++ci) {
      const int ch = tid + ci * 256;
      const int row = ch >> 4, col = (ch & 15) * 8;
      bf16x8 vv = *(const bf16x8*)&V[kb + row * HD_ + col];
#pragma unroll
      for (int e = 0; e < 8; ++e) Vts[(col + e) * 64 + row] = vv[e];
    }
    wait_async0();       // own Q/K asyncs landed (Q covered on first iter)
    __syncthreads();     // everyone's tiles + V transpose landed

    // ---- S = (Q K^T) * SCALE : each wave one 16x16 tile, K-dim 128 ----
    {
      v8f s = {};
#pragma unroll
      for (int ks = 0; ks < 4; ++ks) {
        const int qrow = mt * 16 + l15;
        bf16x8 alo = *(const bf16x8*)&Qs[qrow * 128 + ks * 32 + lh * 8];
        bf16x8 ahi = *(const bf16x8*)&Qs[qrow * 128 + ks * 32 + 16 + lh * 8];
        const int krow = wq * 16 + l15;
        bf16x8 blo = *(const bf16x8*)&Ks[krow * 128 + ks * 32 + lh * 8];
        bf16x8 bhi = *(const bf16x8*)&Ks[krow * 128 + ks * 32 + 16 + lh * 8];
        s = wmma_bf16(join8(alo, ahi), join8(blo, bhi), s);
      }
#pragma unroll
      for (int r = 0; r < 8; ++r) {
        const int row = mt * 16 + r + 8 * lh;
        Ss[row * 64 + wq * 16 + l15] = s[r] * SCALE_;
      }
    }
    __syncthreads();

    // ---- online softmax: wave w owns rows 4w..4w+3, 8 lanes per row ----
    {
      const int row = wave * 4 + (lane >> 3);
      const int kp  = (lane & 7) * 8;          // this lane's 8 keys
      const float mx = rmax[row];
      f32x4 s0 = *(const f32x4*)&Ss[row * 64 + kp];
      f32x4 s1 = *(const f32x4*)&Ss[row * 64 + kp + 4];
      float lm = fmaxf(fmaxf(fmaxf(s0[0], s0[1]), fmaxf(s0[2], s0[3])),
                       fmaxf(fmaxf(s1[0], s1[1]), fmaxf(s1[2], s1[3])));
      lm = fmaxf(lm, __shfl_xor(lm, 1, 8));
      lm = fmaxf(lm, __shfl_xor(lm, 2, 8));
      lm = fmaxf(lm, __shfl_xor(lm, 4, 8));
      const float mnew = fmaxf(mx, lm);
      float p[8], sum = 0.0f;
#pragma unroll
      for (int j = 0; j < 4; ++j) { p[j] = __expf(s0[j] - mnew); sum += p[j]; }
#pragma unroll
      for (int j = 0; j < 4; ++j) { p[4 + j] = __expf(s1[j] - mnew); sum += p[4 + j]; }
      bf16x8 pv;
#pragma unroll
      for (int j = 0; j < 8; ++j) pv[j] = f2bf(p[j]);
      *(bf16x8*)&Pb[row * 64 + kp] = pv;       // one ds_store_b128
      sum += __shfl_xor(sum, 1, 8);
      sum += __shfl_xor(sum, 2, 8);
      sum += __shfl_xor(sum, 4, 8);
      const float scale = __expf(mx - mnew);
      if ((lane & 7) == 0) {
        rmax[row]   = mnew;
        rsum[row]   = rsum[row] * scale + sum;
        rscale[row] = scale;
      }
    }
    __syncthreads();

    // ---- rescale running accumulators ----
#pragma unroll
    for (int r = 0; r < 8; ++r) {
      const float sc = rscale[mt * 16 + r + 8 * lh];
      o[0][r] *= sc; o[1][r] *= sc;
    }
    // ---- O += P @ V : K-dim 64 keys, each wave 16x32 of O ----
#pragma unroll
    for (int ks = 0; ks < 2; ++ks) {
      const int prow = mt * 16 + l15;
      bf16x8 alo = *(const bf16x8*)&Pb[prow * 64 + ks * 32 + lh * 8];
      bf16x8 ahi = *(const bf16x8*)&Pb[prow * 64 + ks * 32 + 16 + lh * 8];
      v16bf a = join8(alo, ahi);
#pragma unroll
      for (int t = 0; t < 2; ++t) {
        const int drow = wq * 32 + t * 16 + l15;
        bf16x8 blo = *(const bf16x8*)&Vts[drow * 64 + ks * 32 + lh * 8];
        bf16x8 bhi = *(const bf16x8*)&Vts[drow * 64 + ks * 32 + 16 + lh * 8];
        o[t] = wmma_bf16(a, join8(blo, bhi), o[t]);
      }
    }
  }

  // ---- normalize and write [b][s][h*HD+d] ----
#pragma unroll
  for (int t = 0; t < 2; ++t)
#pragma unroll
    for (int r = 0; r < 8; ++r) {
      const int row = mt * 16 + r + 8 * lh;
      const int s   = qt * 32 + row;
      const int d   = wq * 32 + t * 16 + l15;
      const float val = o[t][r] / rsum[row];
      Out[(long)(b * S_ + s) * H_ + h * HD_ + d] = f2bf(val);
    }
}

// ---------------------------------------------------------------------------
extern "C" void kernel_launch(void* const* d_in, const int* in_sizes, int n_in,
                              void* d_out, int out_size, void* d_ws, size_t ws_size,
                              hipStream_t stream) {
  (void)in_sizes; (void)n_in; (void)out_size; (void)ws_size;
  const float* X    = (const float*)d_in[0];
  const float* cosp = (const float*)d_in[1];
  const float* sinp = (const float*)d_in[2];
  const float* Wq   = (const float*)d_in[3];
  const float* Wk   = (const float*)d_in[4];
  const float* bk   = (const float*)d_in[5];
  const float* Wv   = (const float*)d_in[6];
  const float* bv   = (const float*)d_in[7];
  const float* Wo   = (const float*)d_in[8];

  const long M  = (long)B_ * S_;     // 4096
  const long KV = (long)NKV_ * HD_;  // 1024

  char* ws = (char*)d_ws;
  size_t off = 0;
  auto wsAlloc = [&](size_t bytes) -> void* {
    void* p = ws + off;
    off = (off + bytes + 255) & ~(size_t)255;
    return p;
  };
  __bf16* Xb   = (__bf16*)wsAlloc((size_t)M * H_ * 2);
  __bf16* Wqb  = (__bf16*)wsAlloc((size_t)H_ * H_ * 2);   // reused for Wo
  __bf16* Wkb  = (__bf16*)wsAlloc((size_t)KV * H_ * 2);
  __bf16* Wvb  = (__bf16*)wsAlloc((size_t)KV * H_ * 2);
  __bf16* qtmp = (__bf16*)wsAlloc((size_t)M * H_ * 2);    // reused for attn
  __bf16* ktmp = (__bf16*)wsAlloc((size_t)M * KV * 2);
  __bf16* vtmp = (__bf16*)wsAlloc((size_t)M * KV * 2);
  __bf16* qhm  = (__bf16*)wsAlloc((size_t)M * H_ * 2);
  __bf16* khm  = (__bf16*)wsAlloc((size_t)M * KV * 2);
  __bf16* vhm  = (__bf16*)wsAlloc((size_t)M * KV * 2);
  __bf16* Wob  = Wqb;   // Wq-bf16 dead after Q projection
  __bf16* attn = qtmp;  // q pre-rope dead after rope_reorder

  dim3 blk(256);
  const long nXH = M * H_ / 4, nWH = (long)H_ * H_ / 4, nWK = KV * H_ / 4;

  cvt_f32_bf16<<<(int)(nXH / 256), blk, 0, stream>>>(X,  Xb,  nXH);
  cvt_f32_bf16<<<(int)(nWH / 256), blk, 0, stream>>>(Wq, Wqb, nWH);
  cvt_f32_bf16<<<(int)(nWK / 256), blk, 0, stream>>>(Wk, Wkb, nWK);
  cvt_f32_bf16<<<(int)(nWK / 256), blk, 0, stream>>>(Wv, Wvb, nWK);

  gemm_bf16<false, false><<<dim3(H_ / 128, M / 128), blk, 0, stream>>>(
      Xb, Wqb, nullptr, qtmp, nullptr, (int)M, H_, H_);
  gemm_bf16<true, false><<<dim3((int)(KV / 128), M / 128), blk, 0, stream>>>(
      Xb, Wkb, bk, ktmp, nullptr, (int)M, (int)KV, H_);
  gemm_bf16<true, false><<<dim3((int)(KV / 128), M / 128), blk, 0, stream>>>(
      Xb, Wvb, bv, vtmp, nullptr, (int)M, (int)KV, H_);

  rope_reorder<<<B_ * S_ * NH_,  64, 0, stream>>>(qtmp, cosp, sinp, qhm, NH_,  1);
  rope_reorder<<<B_ * S_ * NKV_, 64, 0, stream>>>(ktmp, cosp, sinp, khm, NKV_, 1);
  rope_reorder<<<B_ * S_ * NKV_, 64, 0, stream>>>(vtmp, cosp, sinp, vhm, NKV_, 0);

  flash_attn<<<B_ * NH_ * (S_ / 32), blk, 0, stream>>>(qhm, khm, vhm, attn);

  // convert Wo after Wqb is dead, then final projection straight to d_out
  cvt_f32_bf16<<<(int)(nWH / 256), blk, 0, stream>>>(Wo, Wob, nWH);
  gemm_bf16<false, true><<<dim3(H_ / 128, M / 128), blk, 0, stream>>>(
      attn, Wob, nullptr, nullptr, (float*)d_out, (int)M, H_, H_);
}
